// MultiHeadAttention_69784628625810
// MI455X (gfx1250) — matrix operand order repaired and checked by reading, not compiled, not run on hardware
//
#include <hip/hip_runtime.h>
#include <hip/hip_bf16.h>

// Problem constants (match reference)
#define B_   4
#define S_   2048
#define DIN  1024
#define DOUT 1024
#define H_   16
#define HD_  64

typedef __bf16 bf16_t;
typedef __attribute__((ext_vector_type(16))) __bf16 v16bf;
typedef __attribute__((ext_vector_type(8)))  float  v8f;
typedef int v4i __attribute__((vector_size(16)));

// 16x16x32 bf16 fragment: 16 bf16 per lane = 32 bytes = two 16B vector loads.
union Frag { v16bf v; uint4 u[2]; };

__device__ __forceinline__ v8f wmma_bf16(const Frag& a, const Frag& b, v8f c) {
    return __builtin_amdgcn_wmma_f32_16x16x32_bf16(
        /*neg_a=*/false, a.v, /*neg_b=*/false, b.v,
        /*c_mod=*/(short)0, c, /*reuse_a=*/false, /*reuse_b=*/false);
}

// ---- async global->LDS copy (CDNA5 GLOBAL_LOAD_ASYNC_TO_LDS_B128, ASYNCcnt) ----
__device__ __forceinline__ void async_copy_b128(const bf16_t* g, bf16_t* l) {
#if __has_builtin(__builtin_amdgcn_global_load_async_to_lds_b128)
    __builtin_amdgcn_global_load_async_to_lds_b128(
        (__attribute__((address_space(1))) v4i*)g,
        (__attribute__((address_space(3))) v4i*)l,
        /*offset=*/0, /*cpol=*/0);
#else
    *(uint4*)l = *(const uint4*)g;   // synchronous fallback
#endif
}
__device__ __forceinline__ void wait_async_0() { asm volatile("s_wait_asynccnt 0" ::: "memory"); }

// ---------------- cast kernels ----------------
__global__ void cast_f32_bf16(const float* __restrict__ src, bf16_t* __restrict__ dst, int n) {
    int i = blockIdx.x * blockDim.x + threadIdx.x;
    if (i < n) dst[i] = (bf16_t)src[i];
}

// dst[c][r] = src[r][c]  (so GEMM B-fragments read contiguous K)
__global__ void transpose_cast(const float* __restrict__ src, bf16_t* __restrict__ dst,
                               int rows, int cols) {
    int i = blockIdx.x * blockDim.x + threadIdx.x;
    if (i < rows * cols) {
        int r = i / cols, c = i % cols;
        dst[(size_t)c * rows + r] = (bf16_t)src[i];
    }
}

// ---------------- WMMA GEMM with async-LDS weight staging ----------------
// C[m][n] = sum_k A[m][k] * Bt[n][k]  (+ bias for MODE 2)
// MODE 0: bf16 out, [B][H][S][HD]   (Q, K)
// MODE 1: bf16 out, [B][H][HD][S]   (V, transposed for barrier-free attention B-frags)
// MODE 2: f32 out + bias, [M][N]    (output projection; A is ctx in [B][H][S][HD])
// 8 waves/block = 2(M) x 4(N) of 32x32 wave tiles -> block tile 64 x 128.
// Weight tile (128 x 64 K-cols) async-copied to LDS, double-buffered:
//   wait own asynccnt -> one barrier -> issue next chunk (overlaps compute) -> 8 WMMAs.
template<int MODE>
__global__ __launch_bounds__(256)
void gemm_wmma(const bf16_t* __restrict__ A, const bf16_t* __restrict__ Bt,
               const float* __restrict__ bias, void* __restrict__ dstv) {
    const int K = 1024;
    const int KSTEP = 64;
    const int NSTAGE = K / KSTEP;                       // 16
    __shared__ __align__(16) bf16_t ldsB[2][128][KSTEP]; // 2 x 16KB, double buffered

    int wave = threadIdx.x >> 5, lane = threadIdx.x & 31;
    int ln = lane & 15, hi = lane >> 4;
    int m0 = blockIdx.x * 64 + (wave >> 2) * 32;
    int n0base = blockIdx.y * 128;
    int wn = wave & 3;

    // stage the 128 x 64 weight tile at column kc into ldsB[buf]: 1024 x 16B chunks
    auto stage = [&](int kc, int buf) {
#pragma unroll
        for (int j = 0; j < 4; ++j) {
            int cc = (int)threadIdx.x + j * 256;
            int n = cc >> 3, q = cc & 7;                 // 8 x 16B chunks per 64-col row
            async_copy_b128(Bt + (size_t)(n0base + n) * K + kc + q * 8,
                            &ldsB[buf][n][q * 8]);
        }
    };

    stage(0, 0);
    v8f acc[2][2] = {};
#pragma unroll
    for (int i = 0; i < NSTAGE; ++i) {
        wait_async_0();        // our slice of chunk i has landed in LDS
        __syncthreads();       // everyone's slice landed; prior reads of other buffer done
        if (i + 1 < NSTAGE) stage((i + 1) * KSTEP, (i + 1) & 1);  // overlap with compute

#pragma unroll
        for (int ks = 0; ks < 2; ++ks) {
            int kc = i * KSTEP + ks * 32;
            // A fragments direct from global (streamed; rows private to one wave row)
            Frag a[2];
#pragma unroll
            for (int sm = 0; sm < 2; ++sm) {
                const bf16_t* p;
                if (MODE == 2) {
                    int m = m0 + sm * 16 + ln;
                    int bb = m >> 11, s = m & (S_ - 1);
                    int d = kc + hi * 8;       // stays inside one head (32-chunk in 64-head)
                    int h = d >> 6, hd = d & (HD_ - 1);
                    p = A + (((size_t)bb * H_ + h) * S_ + s) * HD_ + hd;
                } else {
                    p = A + (size_t)(m0 + sm * 16 + ln) * K + kc + hi * 8;
                }
                a[sm].u[0] = *(const uint4*)p;
                a[sm].u[1] = *(const uint4*)(p + 16);
            }
            // B fragments from LDS (lane = N column, lane-half selects K 8-groups {hi, hi+2})
            Frag b[2];
#pragma unroll
            for (int sn = 0; sn < 2; ++sn) {
                const bf16_t* p = &ldsB[i & 1][wn * 32 + sn * 16 + ln][ks * 32 + hi * 8];
                b[sn].u[0] = *(const uint4*)p;
                b[sn].u[1] = *(const uint4*)(p + 16);
            }
#pragma unroll
            for (int sm = 0; sm < 2; ++sm)
#pragma unroll
                for (int sn = 0; sn < 2; ++sn)
                    acc[sm][sn] = wmma_bf16(a[sm], b[sn], acc[sm][sn]);
        }
    }

    // C layout: VGPR r -> row (r + hi*8), col = ln
#pragma unroll
    for (int sm = 0; sm < 2; ++sm)
#pragma unroll
        for (int sn = 0; sn < 2; ++sn)
#pragma unroll
            for (int r = 0; r < 8; ++r) {
                int m = m0 + sm * 16 + r + hi * 8;
                int n = n0base + wn * 32 + sn * 16 + ln;
                if (MODE == 2) {
                    ((float*)dstv)[(size_t)m * DOUT + n] = acc[sm][sn][r] + bias[n];
                } else {
                    int bb = m >> 11, s = m & (S_ - 1);
                    int h = n >> 6, hd = n & (HD_ - 1);
                    size_t idx = (MODE == 0)
                        ? (((size_t)bb * H_ + h) * S_ + s) * HD_ + hd
                        : (((size_t)bb * H_ + h) * HD_ + hd) * S_ + s;
                    ((bf16_t*)dstv)[idx] = (bf16_t)acc[sm][sn][r];
                }
            }
}

// ---------------- flash attention (barrier-free) ----------------
// grid: (B*H, S/128); 8 waves/block; wave w -> q rows [q0, q0+16).
// Each wave loops only over its causal key range; V is pre-transposed so
// PV B-fragments are contiguous global loads. P goes through wave-private LDS
// (C-layout -> A-layout) guarded by s_wait_dscnt.
__global__ __launch_bounds__(256)
void attention(const bf16_t* __restrict__ Q, const bf16_t* __restrict__ Kg,
               const bf16_t* __restrict__ Vt, bf16_t* __restrict__ ctx) {
    __shared__ __align__(16) bf16_t ldsP[8][16][32];   // per-wave P staging

    int bh = blockIdx.x;
    int qblk = blockIdx.y;
    int wave = threadIdx.x >> 5, lane = threadIdx.x & 31;
    int ln = lane & 15, hi = lane >> 4;
    int q0 = qblk * 128 + wave * 16;

    const bf16_t* Qp = Q  + (size_t)bh * S_ * HD_;
    const bf16_t* Kp = Kg + (size_t)bh * S_ * HD_;
    const bf16_t* Vp = Vt + (size_t)bh * HD_ * S_;     // [HD][S]

    // Q fragments (A layout), reused across the whole key loop
    Frag qf[2];
#pragma unroll
    for (int c = 0; c < 2; ++c) {
        const bf16_t* p = Qp + (size_t)(q0 + ln) * HD_ + c * 32 + hi * 8;
        qf[c].u[0] = *(const uint4*)p;
        qf[c].u[1] = *(const uint4*)(p + 16);
    }

    v8f O[4] = {};
    float mrow[8], lrow[8];
#pragma unroll
    for (int r = 0; r < 8; ++r) { mrow[r] = -1e30f; lrow[r] = 0.f; }

    const int klim = q0 + 16;          // wave-private causal limit
    for (int kb = 0; kb < klim; kb += 32) {
        if (kb + 32 < klim) {          // prefetch next K/V blocks into cache
            __builtin_prefetch(Kp + (size_t)(kb + 32 + lane) * HD_, 0, 1);
            __builtin_prefetch(Vp + (size_t)(lane * 2) * S_ + kb + 32, 0, 1);
        }

        // S = Q @ K^T for 32 keys: two 16x16 C tiles, hd contraction in 2 chunks
        v8f Sc[2] = {};
#pragma unroll
        for (int t = 0; t < 2; ++t)
#pragma unroll
            for (int c = 0; c < 2; ++c) {
                Frag kf;   // B-matrix: lane = key column, lane-half selects hd 8-groups {hi, hi+2}
                const bf16_t* p = Kp + (size_t)(kb + t * 16 + ln) * HD_ + c * 32 + hi * 8;
                kf.u[0] = *(const uint4*)p;
                kf.u[1] = *(const uint4*)(p + 16);
                Sc[t] = wmma_bf16(qf[c], kf, Sc[t]);
            }

        // online softmax over the 32-key block
        float rowScale[8];
#pragma unroll
        for (int r = 0; r < 8; ++r) {
            int row = q0 + r + hi * 8;
            float a  = Sc[0][r] * 0.125f;   // 1/sqrt(64)
            float b2 = Sc[1][r] * 0.125f;
            if (kb + ln > row)      a  = -1e30f;
            if (kb + 16 + ln > row) b2 = -1e30f;
            float mx = fmaxf(a, b2);
#pragma unroll
            for (int msk = 1; msk < 16; msk <<= 1)
                mx = fmaxf(mx, __shfl_xor(mx, msk, 32));
            float mnew = fmaxf(mrow[r], mx);
            float sc = __expf(mrow[r] - mnew);
            float pa = __expf(a - mnew);
            float pb = __expf(b2 - mnew);
            float rs = pa + pb;
#pragma unroll
            for (int msk = 1; msk < 16; msk <<= 1)
                rs += __shfl_xor(rs, msk, 32);
            lrow[r] = lrow[r] * sc + rs;
            mrow[r] = mnew;
            rowScale[r] = sc;
            // stage P (C-layout element -> row-major LDS) for A-layout reload
            ldsP[wave][r + hi * 8][ln]      = (bf16_t)pa;
            ldsP[wave][r + hi * 8][ln + 16] = (bf16_t)pb;
        }
#pragma unroll
        for (int nc = 0; nc < 4; ++nc)
#pragma unroll
            for (int r = 0; r < 8; ++r) O[nc][r] *= rowScale[r];

        // wave-private LDS round trip: wait only on our own ds stores
        asm volatile("s_wait_dscnt 0" ::: "memory");

        Frag pf;   // P as A-matrix (16 rows x 32 keys)
        {
            const bf16_t* p = &ldsP[wave][ln][hi * 8];
            pf.u[0] = *(const uint4*)p;
            pf.u[1] = *(const uint4*)(p + 16);
        }
#pragma unroll
        for (int nc = 0; nc < 4; ++nc) {
            Frag vf;   // V^T as B-matrix: lane = hd column, keys hi*8..+7 and 16+hi*8..+7
            const bf16_t* p = Vp + (size_t)(nc * 16 + ln) * S_ + kb + hi * 8;
            vf.u[0] = *(const uint4*)p;
            vf.u[1] = *(const uint4*)(p + 16);
            O[nc] = wmma_bf16(pf, vf, O[nc]);
        }
    }

#pragma unroll
    for (int r = 0; r < 8; ++r) {
        float inv = 1.0f / lrow[r];
        int row = q0 + r + hi * 8;
#pragma unroll
        for (int nc = 0; nc < 4; ++nc)
            ctx[((size_t)bh * S_ + row) * HD_ + nc * 16 + ln] = (bf16_t)(O[nc][r] * inv);
    }
}

// ---------------- launch ----------------
extern "C" void kernel_launch(void* const* d_in, const int* in_sizes, int n_in,
                              void* d_out, int out_size, void* d_ws, size_t ws_size,
                              hipStream_t stream) {
    (void)in_sizes; (void)n_in; (void)out_size; (void)ws_size;
    const float* x  = (const float*)d_in[0];
    const float* Wq = (const float*)d_in[1];
    const float* Wk = (const float*)d_in[2];
    const float* Wv = (const float*)d_in[3];
    const float* Wo = (const float*)d_in[4];
    const float* bo = (const float*)d_in[5];
    float* out = (float*)d_out;

    // workspace carve-up (~88 MB total)
    char* ws = (char*)d_ws;
    size_t off = 0;
    auto carve = [&](size_t bytes) -> void* {
        void* p = ws + off;
        off += (bytes + 255) & ~(size_t)255;
        return p;
    };
    const size_t M = (size_t)B_ * S_;
    bf16_t* xbf = (bf16_t*)carve(M * DIN * 2);
    bf16_t* Wtq = (bf16_t*)carve((size_t)DIN * DOUT * 2);
    bf16_t* Wtk = (bf16_t*)carve((size_t)DIN * DOUT * 2);
    bf16_t* Wtv = (bf16_t*)carve((size_t)DIN * DOUT * 2);
    bf16_t* Wob = (bf16_t*)carve((size_t)DOUT * DOUT * 2);
    bf16_t* Qb  = (bf16_t*)carve(M * DOUT * 2);
    bf16_t* Kb  = (bf16_t*)carve(M * DOUT * 2);
    bf16_t* Vb  = (bf16_t*)carve(M * DOUT * 2);   // transposed per-head: [B][H][HD][S]
    bf16_t* ctx = (bf16_t*)carve(M * DOUT * 2);

    int nX = (int)(M * DIN);
    cast_f32_bf16<<<(nX + 255) / 256, 256, 0, stream>>>(x, xbf, nX);
    int nW = DIN * DOUT;
    transpose_cast<<<(nW + 255) / 256, 256, 0, stream>>>(Wq, Wtq, DIN, DOUT);
    transpose_cast<<<(nW + 255) / 256, 256, 0, stream>>>(Wk, Wtk, DIN, DOUT);
    transpose_cast<<<(nW + 255) / 256, 256, 0, stream>>>(Wv, Wtv, DIN, DOUT);
    cast_f32_bf16<<<(nW + 255) / 256, 256, 0, stream>>>(Wo, Wob, nW);

    dim3 gg((unsigned)(M / 64), DOUT / 128);
    gemm_wmma<0><<<gg, 256, 0, stream>>>(xbf, Wtq, nullptr, Qb);
    gemm_wmma<0><<<gg, 256, 0, stream>>>(xbf, Wtk, nullptr, Kb);
    gemm_wmma<1><<<gg, 256, 0, stream>>>(xbf, Wtv, nullptr, Vb);

    attention<<<dim3(B_ * H_, S_ / 128), 256, 0, stream>>>(Qb, Kb, Vb, ctx);

    gemm_wmma<2><<<gg, 256, 0, stream>>>(ctx, Wob, bo, out);
}
